// ALiBiAttention_73581379715523
// MI455X (gfx1250) — matrix operand order repaired and checked by reading, not compiled, not run on hardware
//
#include <hip/hip_runtime.h>

// ---- problem constants (match reference) ----
#define EMBED 1024
#define HEADS 16
#define HEAD_DIM 64
#define SEQT 2048
#define BATCH 2

typedef _Float16 v16h __attribute__((ext_vector_type(16)));
typedef _Float16 v8h  __attribute__((ext_vector_type(8)));
typedef _Float16 h2   __attribute__((ext_vector_type(2)));
typedef float    v8f  __attribute__((ext_vector_type(8)));

__device__ __forceinline__ v8f wmma_f16(v16h a, v16h b, v8f c) {
    return __builtin_amdgcn_wmma_f32_16x16x32_f16(false, a, false, b, (short)0, c, false, false);
}

// A-operand fragment builder: contiguous K mapping (e -> half*16+e)
__device__ __forceinline__ v16h load_afrag(const _Float16* base, int hf) {
    v8h lo = *(const v8h*)(base + hf * 16);
    v8h hi = *(const v8h*)(base + hf * 16 + 8);
    v16h a;
#pragma unroll
    for (int e = 0; e < 8; ++e) { a[e] = lo[e]; a[e + 8] = hi[e]; }
    return a;
}

// ---------------------------------------------------------------------------
// Kernel 1: qkv (T,B,3E) fp32 -> Qh[B,H,T,D]*scale f16, Kh[B,H,T,D] f16,
//           Vt[B,H,D,T] f16 (transposed)
// ---------------------------------------------------------------------------
__global__ void cvt_qkv_kernel(const float* __restrict__ qkv,
                               _Float16* __restrict__ Qh,
                               _Float16* __restrict__ Kh,
                               _Float16* __restrict__ Vt) {
    int idx = blockIdx.x * blockDim.x + threadIdx.x;
    const int N = BATCH * HEADS * SEQT * HEAD_DIM;
    if (idx >= N) return;
    int d = idx & (HEAD_DIM - 1);
    int t = (idx >> 6) & (SEQT - 1);
    int h = (idx >> 17) & (HEADS - 1);
    int b = idx >> 21;
    size_t src = (((size_t)t * BATCH + b) * 3) * EMBED + h * HEAD_DIM + d;
    float q = qkv[src];
    float k = qkv[src + EMBED];
    float v = qkv[src + 2 * EMBED];
    Qh[idx] = (_Float16)(q * 0.125f);   // fold 1/sqrt(64) into Q
    Kh[idx] = (_Float16)k;
    Vt[(((size_t)b * HEADS + h) * HEAD_DIM + d) * SEQT + t] = (_Float16)v;
}

// ---------------------------------------------------------------------------
// Kernel 2: out_w fp32 -> f16
// ---------------------------------------------------------------------------
__global__ void cvt_w_kernel(const float* __restrict__ w, _Float16* __restrict__ wh) {
    int idx = blockIdx.x * blockDim.x + threadIdx.x;
    if (idx < EMBED * EMBED) wh[idx] = (_Float16)w[idx];
}

// ---------------------------------------------------------------------------
// Kernel 3: flash attention, transposed formulation, 32 queries per wave.
//   S^T = K @ Q^T   (softmax rows live in-lane)
//   O^T = V^T @ P^T (correction/normalization are per-lane; no LDS)
// K/V fragments are shared across both 16-query tiles -> 2x data reuse.
// ---------------------------------------------------------------------------
__global__ __launch_bounds__(32)
void attn_kernel(const _Float16* __restrict__ Qh,
                 const _Float16* __restrict__ Kh,
                 const _Float16* __restrict__ Vt,
                 const float*    __restrict__ mask,
                 _Float16*       __restrict__ attnh) {
    const int bid = blockIdx.x;
    const int qt  = bid % (SEQT / 32);
    const int h   = (bid / (SEQT / 32)) % HEADS;
    const int b   = bid / ((SEQT / 32) * HEADS);
    const int lane = threadIdx.x;
    const int hf   = lane >> 4;     // 16-lane half
    const int col  = lane & 15;

    const float alpha = exp2f(-(float)h * (1.0f / (float)HEADS)); // ALIBI_START=1

    const _Float16* Kbh = Kh + ((size_t)(b * HEADS + h)) * SEQT * HEAD_DIM;
    const _Float16* Vbh = Vt + ((size_t)(b * HEADS + h)) * HEAD_DIM * SEQT;

    // per query-tile state (g = 0,1; query = qt*32 + g*16 + col)
    int          q[2];
    const float* Mrow[2];
    v16h         bq[2][2];
    v8f          Oacc[2][4];
    float        m_prev[2], lsum[2];
    const v8f vzero = {0.f, 0.f, 0.f, 0.f, 0.f, 0.f, 0.f, 0.f};

#pragma unroll
    for (int g = 0; g < 2; ++g) {
        q[g] = qt * 32 + g * 16 + col;
        const _Float16* Qbase = Qh + (((size_t)(b * HEADS + h)) * SEQT + q[g]) * HEAD_DIM;
        bq[g][0] = *(const v16h*)(Qbase + hf * 16);
        bq[g][1] = *(const v16h*)(Qbase + 32 + hf * 16);
        Mrow[g]  = mask + ((size_t)b * SEQT + q[g]) * SEQT;
#pragma unroll
        for (int t = 0; t < 4; ++t) Oacc[g][t] = vzero;
        m_prev[g] = -1.0e30f;
        lsum[g]   = 0.0f;
    }

    for (int j = 0; j < SEQT; j += 32) {
        // ---- K A-fragments, shared by both query tiles ----
        v16h ak[2][2];   // [key-frag][d-chunk]
#pragma unroll
        for (int c = 0; c < 2; ++c) {
            ak[0][c] = load_afrag(Kbh + ((size_t)(j + col)) * HEAD_DIM + c * 32, hf);
            ak[1][c] = load_afrag(Kbh + ((size_t)(j + 16 + col)) * HEAD_DIM + c * 32, hf);
        }

        // ---- S^T = K @ Q^T : 8 WMMAs ----
        v8f s[2][2];
#pragma unroll
        for (int g = 0; g < 2; ++g) {
#pragma unroll
            for (int kf = 0; kf < 2; ++kf) {
                v8f acc = wmma_f16(ak[kf][0], bq[g][0], vzero);
                s[g][kf] = wmma_f16(ak[kf][1], bq[g][1], acc);
            }
        }

        // ---- softmax per query tile; build P^T B-fragments ----
        v16h bp[2];
#pragma unroll
        for (int g = 0; g < 2; ++g) {
            v8f mk0 = *(const v8f*)(Mrow[g] + j + 8 * hf);
            v8f mk1 = *(const v8f*)(Mrow[g] + j + 16 + 8 * hf);
            const float t0 = (float)q[g] - (float)(j + 8 * hf);
            const float t1 = t0 - 16.0f;
#pragma unroll
            for (int r = 0; r < 8; ++r) {
                float sc0 = s[g][0][r] - alpha * fabsf(t0 - (float)r) + mk0[r];
                float sc1 = s[g][1][r] - alpha * fabsf(t1 - (float)r) + mk1[r];
                s[g][0][r] = fmaxf(sc0, -10000.0f);
                s[g][1][r] = fmaxf(sc1, -10000.0f);
            }

            float mx = fmaxf(s[g][0][0], s[g][1][0]);
#pragma unroll
            for (int r = 1; r < 8; ++r) mx = fmaxf(mx, fmaxf(s[g][0][r], s[g][1][r]));
            mx = fmaxf(mx, __shfl_xor(mx, 16, 32));

            const float mn   = fmaxf(m_prev[g], mx);
            const float corr = __expf(m_prev[g] - mn);
            m_prev[g] = mn;

            float p0[8], p1[8];
            float rs = 0.0f;
#pragma unroll
            for (int r = 0; r < 8; ++r) {
                p0[r] = __expf(s[g][0][r] - mn);
                p1[r] = __expf(s[g][1][r] - mn);
                rs += p0[r] + p1[r];
            }
            rs += __shfl_xor(rs, 16, 32);
            lsum[g] = lsum[g] * corr + rs;

#pragma unroll
            for (int t = 0; t < 4; ++t) Oacc[g][t] = Oacc[g][t] * corr;

            // packed f16x2 cross-half exchange: 8 shuffles per tile
#pragma unroll
            for (int r = 0; r < 8; ++r) {
                h2 own;
                own.x = (_Float16)p0[r];
                own.y = (_Float16)p1[r];
                int ownu  = __builtin_bit_cast(int, own);
                int partu = __shfl_xor(ownu, 16, 32);
                h2 part   = __builtin_bit_cast(h2, partu);
                bp[g][r]     = hf ? part.y : own.x;   // keys hf*16 + r
                bp[g][8 + r] = hf ? own.y  : part.x;  // keys hf*16 + 8 + r
            }
        }

        // ---- O^T += V^T @ P^T : V fragment shared by both tiles, 8 WMMAs ----
#pragma unroll
        for (int t = 0; t < 4; ++t) {
            v16h av = load_afrag(Vbh + ((size_t)(t * 16 + col)) * SEQT + j, hf);
#pragma unroll
            for (int g = 0; g < 2; ++g) Oacc[g][t] = wmma_f16(av, bp[g], Oacc[g][t]);
        }
    }

    // ---- normalize (per-lane) and store: contiguous 16B f16 stores ----
#pragma unroll
    for (int g = 0; g < 2; ++g) {
        const float inv = 1.0f / lsum[g];
        _Float16* obase = attnh + ((size_t)q[g] * BATCH + b) * EMBED + h * HEAD_DIM + 8 * hf;
#pragma unroll
        for (int t = 0; t < 4; ++t) {
            v8h st;
#pragma unroll
            for (int r = 0; r < 8; ++r) st[r] = (_Float16)(Oacc[g][t][r] * inv);
            *(v8h*)(obase + t * 16) = st;
        }
    }
}

// ---------------------------------------------------------------------------
// Kernel 4: projection GEMM  out(4096,1024) = attn(4096,1024) @ W^T + bias
// One wave per 16x64 output tile (A-fragment reused across 4 N-tiles).
// ---------------------------------------------------------------------------
__global__ __launch_bounds__(128)
void proj_kernel(const _Float16* __restrict__ Ah,
                 const _Float16* __restrict__ Wh,
                 const float*    __restrict__ bias,
                 float*          __restrict__ out) {
    const int wave = blockIdx.x * (blockDim.x >> 5) + (threadIdx.x >> 5);
    const int lane = threadIdx.x & 31;
    const int hf   = lane >> 4;
    const int col  = lane & 15;
    const int mt   = wave >> 4;   // 256 row tiles
    const int nt   = wave & 15;   // 16 column super-tiles of 64

    const _Float16* Abase = Ah + ((size_t)(mt * 16 + col)) * EMBED;

    v8f acc[4];
    const v8f vzero = {0.f, 0.f, 0.f, 0.f, 0.f, 0.f, 0.f, 0.f};
#pragma unroll
    for (int t = 0; t < 4; ++t) acc[t] = vzero;

    for (int kc = 0; kc < EMBED; kc += 32) {
        v16h a = load_afrag(Abase + kc, hf);
#pragma unroll
        for (int t = 0; t < 4; ++t) {
            v16h bw = *(const v16h*)(Wh + ((size_t)(nt * 64 + t * 16 + col)) * EMBED + kc + hf * 16);
            acc[t] = wmma_f16(a, bw, acc[t]);
        }
    }

#pragma unroll
    for (int r = 0; r < 8; ++r) {
        const int gm = mt * 16 + r + 8 * hf;
#pragma unroll
        for (int t = 0; t < 4; ++t) {
            const int gn = nt * 64 + t * 16 + col;
            out[(size_t)gm * EMBED + gn] = acc[t][r] + bias[gn];
        }
    }
}

// ---------------------------------------------------------------------------
extern "C" void kernel_launch(void* const* d_in, const int* in_sizes, int n_in,
                              void* d_out, int out_size, void* d_ws, size_t ws_size,
                              hipStream_t stream) {
    const float* qkv   = (const float*)d_in[0];
    const float* mask  = (const float*)d_in[1];
    const float* out_w = (const float*)d_in[2];
    const float* out_b = (const float*)d_in[3];
    float* out = (float*)d_out;

    const size_t NQ = (size_t)BATCH * HEADS * SEQT * HEAD_DIM; // 4M elems
    char* ws = (char*)d_ws;
    _Float16* Qh    = (_Float16*)(ws);
    _Float16* Kh    = (_Float16*)(ws + 2 * NQ);
    _Float16* Vt    = (_Float16*)(ws + 4 * NQ);
    _Float16* attnh = (_Float16*)(ws + 6 * NQ);
    _Float16* Wh    = (_Float16*)(ws + 8 * NQ);   // total 34MB

    {   // qkv -> Q/K/V f16 (Q prescaled, V transposed)
        const int N = (int)NQ;
        cvt_qkv_kernel<<<(N + 255) / 256, 256, 0, stream>>>(qkv, Qh, Kh, Vt);
    }
    {   // out_w -> f16
        const int N = EMBED * EMBED;
        cvt_w_kernel<<<(N + 255) / 256, 256, 0, stream>>>(out_w, Wh);
    }
    {   // flash attention: one wave per (b,h,32-query tile)
        const int blocks = BATCH * HEADS * (SEQT / 32);   // 2048
        attn_kernel<<<blocks, 32, 0, stream>>>(Qh, Kh, Vt, mask, attnh);
    }
    {   // projection: 4096 waves, 4 waves/block
        const int waves = (SEQT * BATCH / 16) * (EMBED / 64); // 4096
        proj_kernel<<<waves / 4, 128, 0, stream>>>(attnh, Wh, out_b, out);
    }
}
